// NCF_1675037246099
// MI455X (gfx1250) — compile-verified, matrix operand order checked by reading, not run on hardware
//
#include <hip/hip_runtime.h>
#include <hip/hip_bf16.h>

// ---------------------------------------------------------------------------
// NCF-style dual scoring head for MI455X (gfx1250, wave32, WMMA).
// Heavy GEMMs run on v_wmma_f32_16x16x32_bf16 with f32 accumulation.
// Pair-score kernel uses 64x64 wave tiles (16 WMMA : 16 b128 loads per
// K-step) so the hot loop is WMMA-issue-bound, with all operands L2-resident.
// ---------------------------------------------------------------------------

typedef __attribute__((ext_vector_type(16))) __bf16        bf16x16;
typedef __attribute__((ext_vector_type(8)))  float         f32x8;
typedef __attribute__((ext_vector_type(4)))  unsigned int  u32x4;

#define DD  128   // embed dim
#define NN  4096  // cells
#define MM  8192  // genes

__device__ __forceinline__ unsigned short f2bf_rne(float x) {
  unsigned int u = __float_as_uint(x);
  unsigned int r = (u + 0x7FFFu + ((u >> 16) & 1u)) >> 16;
  return (unsigned short)r;
}

union FragBF {
  bf16x16 v;
  u32x4   q[2];
};

// ----------------------------- small helpers -------------------------------

__global__ void cvt_f32_bf16(const float* __restrict__ in,
                             unsigned short* __restrict__ out, int n) {
  int i = blockIdx.x * blockDim.x + threadIdx.x;
  if (i < n) out[i] = f2bf_rne(in[i]);
}

// out[c*rows + r] = bf16(in[r*cols + c])   (store W^T so GEMM B-frag loads are
// K-contiguous, same addressing as A-frags)
__global__ void transpose_f32_bf16(const float* __restrict__ in,
                                   unsigned short* __restrict__ out,
                                   int rows, int cols) {
  int i = blockIdx.x * blockDim.x + threadIdx.x;
  if (i < rows * cols) {
    int r = i / cols, c = i % cols;
    out[(size_t)c * rows + r] = f2bf_rne(in[i]);
  }
}

// eu = u @ Wb (K small, VALU), also emit bf16(eu * w_prod) for the pair GEMM.
__global__ void embed_u_kernel(const float* __restrict__ u,
                               const float* __restrict__ Wb,
                               const float* __restrict__ w_prod,
                               float* __restrict__ euF,
                               unsigned short* __restrict__ euwB, int K) {
  int n = blockIdx.x;
  int d = threadIdx.x;            // 0..127
  const float* up = u + (size_t)n * K;
  float acc = 0.f;
  for (int k = 0; k < K; ++k) acc = fmaf(up[k], Wb[(size_t)k * DD + d], acc);
  size_t idx = (size_t)n * DD + d;
  euF[idx]  = acc;
  euwB[idx] = f2bf_rne(acc * w_prod[d]);
}

// out[row] = dot(X[row, 0:128], w[0:128])   (rank-1 additive terms)
__global__ void rowdot_kernel(const float* __restrict__ X,
                              const float* __restrict__ w,
                              float* __restrict__ out) {
  __shared__ float red[4];
  int row = blockIdx.x, t = threadIdx.x;  // 128 threads
  float p = X[(size_t)row * DD + t] * w[t];
  for (int off = 16; off > 0; off >>= 1) p += __shfl_xor(p, off, 32);
  if ((t & 31) == 0) red[t >> 5] = p;
  __syncthreads();
  if (t == 0) out[row] = red[0] + red[1] + red[2] + red[3];
}

// --------------------- generic bf16 WMMA GEMM ------------------------------
// C[rows x cols] = act(A[rows x K] @ BT[cols x K]^T)
// Block = 256 thr = 8 waves (2 row-groups x 4 col-groups); wave tile 64x32.
__global__ __launch_bounds__(256)
void gemm_bf16_kernel(const unsigned short* __restrict__ A,
                      const unsigned short* __restrict__ BT,
                      int K, int cols,
                      unsigned short* __restrict__ outB,
                      float* __restrict__ outF, int do_relu) {
  const int lane = threadIdx.x & 31;
  const int wave = threadIdx.x >> 5;
  const int wr = wave >> 2, wc = wave & 3;
  const int lr = lane & 15, kh = lane >> 4;
  const size_t r0 = (size_t)blockIdx.y * 128 + (size_t)wr * 64;
  const size_t c0 = (size_t)blockIdx.x * 128 + (size_t)wc * 32;

  f32x8 acc[4][2] = {};

  for (int kk = 0; kk < K; kk += 32) {
    FragBF fa[4], fb[2];
#pragma unroll
    for (int i = 0; i < 4; ++i) {
      const unsigned short* p = A + ((r0 + 16 * i + lr) * (size_t)K + kk + 8 * kh);
      fa[i].q[0] = *(const u32x4*)p;
      fa[i].q[1] = *(const u32x4*)(p + 16);
    }
#pragma unroll
    for (int j = 0; j < 2; ++j) {
      const unsigned short* p = BT + ((c0 + 16 * j + lr) * (size_t)K + kk + 8 * kh);
      fb[j].q[0] = *(const u32x4*)p;
      fb[j].q[1] = *(const u32x4*)(p + 16);
    }
#pragma unroll
    for (int i = 0; i < 4; ++i)
#pragma unroll
      for (int j = 0; j < 2; ++j)
        acc[i][j] = __builtin_amdgcn_wmma_f32_16x16x32_bf16(
            false, fa[i].v, false, fb[j].v, (short)0, acc[i][j], false, false);
  }

#pragma unroll
  for (int i = 0; i < 4; ++i)
#pragma unroll
    for (int j = 0; j < 2; ++j)
#pragma unroll
      for (int v = 0; v < 8; ++v) {
        size_t row = r0 + 16 * i + v + 8 * kh;
        size_t col = c0 + 16 * j + lr;
        float x = acc[i][j][v];
        if (do_relu) x = fmaxf(x, 0.f);
        size_t idx = row * (size_t)cols + col;
        if (outF) outF[idx] = x;
        if (outB) outB[idx] = f2bf_rne(x);
      }
}

// --------------------- pair-score WMMA GEMM --------------------------------
// out[n,m] = act( (euw @ ev^T)[n,m] + r[n] + c[m] + bias ), K = 128 fixed.
// Wave tile 64x64 (16 accumulators); WG tile 128 x 256 (8 waves, 2x4).
__global__ __launch_bounds__(256)
void pair_score_kernel(const unsigned short* __restrict__ Abf,  // [NN,128] eu*w_prod
                       const unsigned short* __restrict__ Bbf,  // [MM,128] ev
                       const float* __restrict__ rterm,
                       const float* __restrict__ cterm,
                       const float* __restrict__ biasp,
                       float* __restrict__ out, int act_sigmoid) {
  const int lane = threadIdx.x & 31;
  const int wave = threadIdx.x >> 5;
  const int wr = wave >> 2, wc = wave & 3;
  const int lr = lane & 15, kh = lane >> 4;
  const size_t r0 = (size_t)blockIdx.y * 128 + (size_t)wr * 64;  // cell rows
  const size_t c0 = (size_t)blockIdx.x * 256 + (size_t)wc * 64;  // gene cols
  const float bias = biasp[0];

  f32x8 acc[4][4] = {};

#pragma unroll
  for (int kk = 0; kk < DD; kk += 32) {
    FragBF fa[4], fb[4];
#pragma unroll
    for (int i = 0; i < 4; ++i) {
      const unsigned short* p = Abf + ((r0 + 16 * i + lr) * (size_t)DD + kk + 8 * kh);
      fa[i].q[0] = *(const u32x4*)p;
      fa[i].q[1] = *(const u32x4*)(p + 16);
    }
#pragma unroll
    for (int j = 0; j < 4; ++j) {
      const unsigned short* p = Bbf + ((c0 + 16 * j + lr) * (size_t)DD + kk + 8 * kh);
      fb[j].q[0] = *(const u32x4*)p;
      fb[j].q[1] = *(const u32x4*)(p + 16);
    }
#pragma unroll
    for (int i = 0; i < 4; ++i)
#pragma unroll
      for (int j = 0; j < 4; ++j)
        acc[i][j] = __builtin_amdgcn_wmma_f32_16x16x32_bf16(
            false, fa[i].v, false, fb[j].v, (short)0, acc[i][j], false, false);
  }

#pragma unroll
  for (int j = 0; j < 4; ++j) {
    float cc = cterm[c0 + 16 * j + lr];
#pragma unroll
    for (int i = 0; i < 4; ++i)
#pragma unroll
      for (int v = 0; v < 8; ++v) {
        size_t row = r0 + 16 * i + v + 8 * kh;
        float x = acc[i][j][v] + rterm[row] + cc + bias;
        if (act_sigmoid)
          x = 1.f / (1.f + __expf(-x));
        else
          x = fmaxf(x, 0.f);
        out[row * (size_t)MM + c0 + 16 * j + lr] = x;
      }
  }
}

// ---------------------------------------------------------------------------

extern "C" void kernel_launch(void* const* d_in, const int* in_sizes, int n_in,
                              void* d_out, int out_size, void* d_ws, size_t ws_size,
                              hipStream_t stream) {
  (void)in_sizes; (void)n_in; (void)out_size; (void)ws_size;
  const float* u1   = (const float*)d_in[0];
  const float* u2   = (const float*)d_in[1];
  const float* v1   = (const float*)d_in[2];
  const float* v2   = (const float*)d_in[3];
  const float* Wb1  = (const float*)d_in[4];
  const float* Wb2  = (const float*)d_in[5];
  const float* Wrna1= (const float*)d_in[6];   // [256, 256]
  const float* Wrna2= (const float*)d_in[7];   // [256, 128]
  const float* gact = (const float*)d_in[8];   // [512, 256]
  const float* wp1  = (const float*)d_in[9];
  const float* wa1  = (const float*)d_in[10];  // [256]
  const float* wp2  = (const float*)d_in[11];
  const float* wa2  = (const float*)d_in[12];
  const float* b1   = (const float*)d_in[13];
  const float* b2   = (const float*)d_in[14];

  // output regions (return order: out1, out2, eu1, ev1, eu2, ev2)
  float* out1 = (float*)d_out;
  float* out2 = out1 + (size_t)NN * MM;
  float* eu1  = out2 + (size_t)NN * MM;
  float* ev1  = eu1 + (size_t)NN * DD;
  float* eu2  = ev1 + (size_t)MM * DD;
  float* ev2  = eu2 + (size_t)NN * DD;

  // workspace carve-up (256B aligned)
  char* ws = (char*)d_ws;
  size_t off = 0;
  auto carve = [&](size_t bytes) {
    char* p = ws + off;
    off = (off + bytes + 255) & ~(size_t)255;
    return p;
  };
  unsigned short* euw1 = (unsigned short*)carve((size_t)NN * DD * 2);
  unsigned short* euw2 = (unsigned short*)carve((size_t)NN * DD * 2);
  unsigned short* ev1b = (unsigned short*)carve((size_t)MM * DD * 2);
  unsigned short* ev2b = (unsigned short*)carve((size_t)MM * DD * 2);
  float*          r1   = (float*)carve((size_t)NN * 4);
  float*          r2   = (float*)carve((size_t)NN * 4);
  float*          c1   = (float*)carve((size_t)MM * 4);
  float*          c2   = (float*)carve((size_t)MM * 4);
  unsigned short* v1b  = (unsigned short*)carve((size_t)MM * 256 * 2);
  unsigned short* v2b  = (unsigned short*)carve((size_t)MM * 512 * 2);
  unsigned short* W1T  = (unsigned short*)carve((size_t)256 * 256 * 2); // [256 cols][256 K]
  unsigned short* W2T  = (unsigned short*)carve((size_t)128 * 256 * 2); // [128 cols][256 K]
  unsigned short* GAT  = (unsigned short*)carve((size_t)256 * 512 * 2); // [256 cols][512 K]
  unsigned short* h1b  = (unsigned short*)carve((size_t)MM * 256 * 2);
  unsigned short* gb   = (unsigned short*)carve((size_t)MM * 256 * 2);
  unsigned short* h2b  = (unsigned short*)carve((size_t)MM * 256 * 2);

  // 1) conversions / weight transposes
  {
    int n = MM * 256;
    cvt_f32_bf16<<<(n + 255) / 256, 256, 0, stream>>>(v1, v1b, n);
    n = MM * 512;
    cvt_f32_bf16<<<(n + 255) / 256, 256, 0, stream>>>(v2, v2b, n);
    n = 256 * 256;
    transpose_f32_bf16<<<(n + 255) / 256, 256, 0, stream>>>(Wrna1, W1T, 256, 256);
    n = 256 * 128;
    transpose_f32_bf16<<<(n + 255) / 256, 256, 0, stream>>>(Wrna2, W2T, 256, 128);
    n = 512 * 256;
    transpose_f32_bf16<<<(n + 255) / 256, 256, 0, stream>>>(gact, GAT, 512, 256);
  }

  // 2) cell embeddings + row terms
  embed_u_kernel<<<NN, 128, 0, stream>>>(u1, Wb1, wp1, eu1, euw1, 100);
  embed_u_kernel<<<NN, 128, 0, stream>>>(u2, Wb2, wp2, eu2, euw2, 100);
  rowdot_kernel<<<NN, 128, 0, stream>>>(eu1, wa1, r1);
  rowdot_kernel<<<NN, 128, 0, stream>>>(eu2, wa2, r2);

  // 3) gene embeddings (WMMA chain)
  // h1 = relu(v1 @ Wrna1)               [8192,256]
  gemm_bf16_kernel<<<dim3(2, MM / 128), 256, 0, stream>>>(v1b, W1T, 256, 256, h1b, nullptr, 1);
  // ev1 = h1 @ Wrna2                    [8192,128]
  gemm_bf16_kernel<<<dim3(1, MM / 128), 256, 0, stream>>>(h1b, W2T, 256, 128, ev1b, ev1, 0);
  rowdot_kernel<<<MM, 128, 0, stream>>>(ev1, wa1 + DD, c1);
  // g = v2 @ gene_act                   [8192,256]
  gemm_bf16_kernel<<<dim3(2, MM / 128), 256, 0, stream>>>(v2b, GAT, 512, 256, gb, nullptr, 0);
  // h2 = relu(g @ Wrna1)                [8192,256]
  gemm_bf16_kernel<<<dim3(2, MM / 128), 256, 0, stream>>>(gb, W1T, 256, 256, h2b, nullptr, 1);
  // ev2 = h2 @ Wrna2                    [8192,128]
  gemm_bf16_kernel<<<dim3(1, MM / 128), 256, 0, stream>>>(h2b, W2T, 256, 128, ev2b, ev2, 0);
  rowdot_kernel<<<MM, 128, 0, stream>>>(ev2, wa2 + DD, c2);

  // 4) all-pairs scores (dominant WMMA work, fused epilogues)
  pair_score_kernel<<<dim3(MM / 256, NN / 128), 256, 0, stream>>>(
      euw1, ev1b, r1, c1, b1, out1, 0);
  pair_score_kernel<<<dim3(MM / 256, NN / 128), 256, 0, stream>>>(
      euw2, ev2b, r2, c2, b2, out2, 1);
}
